// MyMSA_70282844832027
// MI455X (gfx1250) — matrix operand order repaired and checked by reading, not compiled, not run on hardware
//
#include <hip/hip_runtime.h>
#include <hip/hip_bf16.h>

#define NB 8
#define NH 12
#define SEQ 1024
#define DHEAD 64
#define DMODEL 768

typedef __attribute__((ext_vector_type(16))) __bf16 bf16x16;
typedef __attribute__((ext_vector_type(8)))  __bf16 bf16x8;
typedef __attribute__((ext_vector_type(8)))  float  f32x8;

union Frag {
  bf16x16 v;
  struct { bf16x8 lo; bf16x8 hi; } p;
};

__device__ __forceinline__ f32x8 wmma_bf16(bf16x16 a, bf16x16 b, f32x8 c) {
  // D = A(16x32 bf16) * B(32x16 bf16) + C(16x16 f32)
  return __builtin_amdgcn_wmma_f32_16x16x32_bf16(
      /*neg_a=*/false, a, /*neg_b=*/false, b,
      /*c_mod=*/(short)0, c, /*reuse_a=*/false, /*reuse_b=*/false);
}

// ---------------------------------------------------------------------------
// Kernel 1: per-head Q/K/V projection, f32 -> bf16, WMMA GEMM.
//   Qb : [B*H][S][DH] bf16   (Q pre-scaled by 1/sqrt(DH)*log2(e), bias incl.)
//   Kb : [B*H][S][DH] bf16
//   Vt : [B*H][DH][S] bf16   (V transposed for contiguous B-frag loads)
// One wave handles 16 rows of S for one (b,h).
// ---------------------------------------------------------------------------
__global__ __launch_bounds__(256) void qkv_proj_kernel(
    const float* __restrict__ x,
    const float* __restrict__ Wq, const float* __restrict__ Wk,
    const float* __restrict__ Wv,
    const float* __restrict__ bq, const float* __restrict__ bk,
    const float* __restrict__ bv,
    __bf16* __restrict__ Qb, __bf16* __restrict__ Kb, __bf16* __restrict__ Vt)
{
  const int lane = threadIdx.x & 31;
  const int w    = threadIdx.x >> 5;
  const int n    = lane & 15;
  const int g    = lane >> 4;
  const int wgid  = blockIdx.x * 8 + w;
  const int stile = wgid & (SEQ / 16 - 1);
  const int bh    = wgid >> 6;
  const int h     = bh % NH;
  const int b     = bh / NH;
  const int s0    = stile * 16;

  // A fragments: rows of x (m = lane&15), k = head-feature d.
  // k order used here (B matches): elems 0..15 -> k = 16g..16g+15 contiguous
  const float* xrow = x + ((size_t)(b * SEQ + s0 + n)) * DMODEL + h * DHEAD;
  Frag A[2];
#pragma unroll
  for (int ks = 0; ks < 2; ++ks) {
    bf16x8 lo, hi;
    const float* plo = xrow + 32 * ks + 16 * g;
    const float* phi = xrow + 32 * ks + 16 * g + 8;
#pragma unroll
    for (int i = 0; i < 8; ++i) {
      lo[i] = (__bf16)plo[i];
      hi[i] = (__bf16)phi[i];
    }
    A[ks].p.lo = lo;
    A[ks].p.hi = hi;
  }

  const float QSCALE = 0.125f * 1.44269504088896340736f; // 1/sqrt(64)*log2(e)

  __bf16* qdst = Qb + (size_t)bh * SEQ * DHEAD;
  __bf16* kdst = Kb + (size_t)bh * SEQ * DHEAD;
  __bf16* vdst = Vt + (size_t)bh * DHEAD * SEQ;

#pragma unroll
  for (int et = 0; et < 4; ++et) {
    const int e = 16 * et + n; // this lane's output column
    const float* wqrow = Wq + ((size_t)h * DHEAD + e) * DHEAD;
    const float* wkrow = Wk + ((size_t)h * DHEAD + e) * DHEAD;
    const float* wvrow = Wv + ((size_t)h * DHEAD + e) * DHEAD;

    f32x8 cq = {}, ck = {}, cv = {};
#pragma unroll
    for (int ks = 0; ks < 2; ++ks) {
      // B frag: lane holds column n, k = 16g..16g+15 contiguous (same order as A).
      Frag Bq_, Bk_, Bv_;
      const int off = 32 * ks + 16 * g;
#pragma unroll
      for (int j = 0; j < 8; ++j) {
        Bq_.p.lo[j] = (__bf16)wqrow[off + j];
        Bq_.p.hi[j] = (__bf16)wqrow[off + 8 + j];
        Bk_.p.lo[j] = (__bf16)wkrow[off + j];
        Bk_.p.hi[j] = (__bf16)wkrow[off + 8 + j];
        Bv_.p.lo[j] = (__bf16)wvrow[off + j];
        Bv_.p.hi[j] = (__bf16)wvrow[off + 8 + j];
      }
      cq = wmma_bf16(A[ks].v, Bq_.v, cq);
      ck = wmma_bf16(A[ks].v, Bk_.v, ck);
      cv = wmma_bf16(A[ks].v, Bv_.v, cv);
    }

    const float bqv = bq[h * DHEAD + e];
    const float bkv = bk[h * DHEAD + e];
    const float bvv = bv[h * DHEAD + e];

    // C layout: vgpr i -> row m = i + 8g, column = n (fixed per lane)
#pragma unroll
    for (int i = 0; i < 8; ++i) {
      const int s = s0 + i + 8 * g;
      qdst[(size_t)s * DHEAD + e] = (__bf16)((cq[i] + bqv) * QSCALE);
      kdst[(size_t)s * DHEAD + e] = (__bf16)(ck[i] + bkv);
    }
    // V transposed: Vt[d][t]; 8 consecutive t per lane -> one b128 store
    bf16x8 vpk;
#pragma unroll
    for (int i = 0; i < 8; ++i) vpk[i] = (__bf16)(cv[i] + bvv);
    *(bf16x8*)(vdst + (size_t)e * SEQ + s0 + 8 * g) = vpk;
  }
}

// ---------------------------------------------------------------------------
// Kernel 2: flash-attention over t (tiles of 32).
// Scores computed TRANSPOSED (S^T = K * Q^T) so each lane's accumulator
// components all belong to one query row: row-max needs only a per-lane
// reduction + one shfl_xor(16); corr needs a single exp2; the P tile packs
// into two ds_store_b128. P.V and the ones-matrix row-sum use WMMA.
// ---------------------------------------------------------------------------
__global__ __launch_bounds__(256) void attn_kernel(
    const __bf16* __restrict__ Qb, const __bf16* __restrict__ Kb,
    const __bf16* __restrict__ Vt, float* __restrict__ out)
{
  __shared__ __bf16 Plds[8][16][32]; // per-wave 16(s)x32(t) P tile, 8 KB

  const int lane = threadIdx.x & 31;
  const int w    = threadIdx.x >> 5;
  const int n    = lane & 15;
  const int g    = lane >> 4;
  const int wgid  = blockIdx.x * 8 + w;
  const int stile = wgid & (SEQ / 16 - 1);
  const int bh    = wgid >> 6;
  const int h     = bh % NH;
  const int b     = bh / NH;
  const int s0    = stile * 16;

  // Q as loop-invariant B-fragments: lane holds column n = s, k = e contiguous
  const __bf16* qrow = Qb + ((size_t)bh * SEQ + s0 + n) * DHEAD;
  Frag Bq[2];
#pragma unroll
  for (int ks = 0; ks < 2; ++ks) {
    Bq[ks].p.lo = *(const bf16x8*)(qrow + 32 * ks + 16 * g);
    Bq[ks].p.hi = *(const bf16x8*)(qrow + 32 * ks + 16 * g + 8);
  }

  // All-ones B fragment: rowsum(P) = P x ones, replicated across lanes.
  Frag Bones;
#pragma unroll
  for (int j = 0; j < 8; ++j) {
    Bones.p.lo[j] = (__bf16)1.0f;
    Bones.p.hi[j] = (__bf16)1.0f;
  }

  const __bf16* kbase = Kb + (size_t)bh * SEQ * DHEAD;
  const __bf16* vbase = Vt + (size_t)bh * DHEAD * SEQ;

  f32x8 O[4] = {};
  f32x8 lrun = {};          // component layout: slot i -> row s0 + i + 8g
  float mrun = -3.0e38f;    // lane layout: this lane's row s0 + n

  for (int t0 = 0; t0 < SEQ; t0 += 32) {
    // ---- transposed scores: two 16(t)x16(s) tiles ----
    f32x8 sc[2];
    sc[0] = f32x8{};
    sc[1] = f32x8{};
#pragma unroll
    for (int tt = 0; tt < 2; ++tt) {
      const __bf16* krow = kbase + (size_t)(t0 + 16 * tt + n) * DHEAD;
#pragma unroll
      for (int ks = 0; ks < 2; ++ks) {
        Frag Ak;
        Ak.p.lo = *(const bf16x8*)(krow + 32 * ks + 16 * g);
        Ak.p.hi = *(const bf16x8*)(krow + 32 * ks + 16 * g + 8);
        sc[tt] = wmma_bf16(Ak.v, Bq[ks].v, sc[tt]);
      }
    }

    // ---- online softmax (log2 domain; Q pre-scaled) ----
    // All 16 components of this lane belong to row s = s0 + n.
    float mloc = fmaxf(sc[0][0], sc[1][0]);
#pragma unroll
    for (int i = 1; i < 8; ++i)
      mloc = fmaxf(mloc, fmaxf(sc[0][i], sc[1][i]));
    mloc = fmaxf(mloc, __shfl_xor(mloc, 16, 32));  // merge the two lane halves
    const float mnew = fmaxf(mrun, mloc);
    const float corr = __builtin_exp2f(mrun - mnew);
    mrun = mnew;

    // ---- exp, pack, store P tile to LDS (contiguous along t per lane) ----
#pragma unroll
    for (int tt = 0; tt < 2; ++tt) {
      bf16x8 pk;
#pragma unroll
      for (int i = 0; i < 8; ++i)
        pk[i] = (__bf16)__builtin_exp2f(sc[tt][i] - mrun);
      *(bf16x8*)(&Plds[w][n][16 * tt + 8 * g]) = pk;
    }
    // Same-wave DS ops complete in order; barrier prevents compiler reorder.
    asm volatile("s_wait_dscnt 0" ::: "memory");

    // ---- P A-fragment (rows m = s, k = t) ----
    Frag Ap;
    const __bf16* prow = &Plds[w][n][0];
    Ap.p.lo = *(const bf16x8*)(prow + 16 * g);
    Ap.p.hi = *(const bf16x8*)(prow + 16 * g + 8);

    // ---- row-sum via matrix core: sl[m] = sum_k P[m][k] ----
    f32x8 zero = {};
    f32x8 sl = wmma_bf16(Ap.v, Bones.v, zero);

    // corr (lane layout, row = n) -> component layout (row = i + 8g)
    f32x8 corr_c;
#pragma unroll
    for (int i = 0; i < 8; ++i) corr_c[i] = __shfl(corr, i + 8 * g, 32);
#pragma unroll
    for (int i = 0; i < 8; ++i) lrun[i] = lrun[i] * corr_c[i] + sl[i];
#pragma unroll
    for (int dt = 0; dt < 4; ++dt)
#pragma unroll
      for (int i = 0; i < 8; ++i) O[dt][i] *= corr_c[i];

    // ---- O += P(16x32) * V(32x64): 4 WMMAs against V^T rows ----
#pragma unroll
    for (int dt = 0; dt < 4; ++dt) {
      const __bf16* vrow = vbase + (size_t)(16 * dt + n) * SEQ + t0 + 16 * g;
      Frag Bv_;
      Bv_.p.lo = *(const bf16x8*)(vrow);
      Bv_.p.hi = *(const bf16x8*)(vrow + 8);
      O[dt] = wmma_bf16(Ap.v, Bv_.v, O[dt]);
    }
  }

  // ---- epilogue: normalize and write f32 output [B][S][H*DH] ----
  f32x8 rl;
#pragma unroll
  for (int i = 0; i < 8; ++i) rl[i] = 1.0f / lrun[i];
#pragma unroll
  for (int dt = 0; dt < 4; ++dt)
#pragma unroll
    for (int i = 0; i < 8; ++i) {
      const int s = s0 + i + 8 * g;
      out[((size_t)(b * SEQ + s)) * DMODEL + h * DHEAD + 16 * dt + n] =
          O[dt][i] * rl[i];
    }
}

extern "C" void kernel_launch(void* const* d_in, const int* in_sizes, int n_in,
                              void* d_out, int out_size, void* d_ws,
                              size_t ws_size, hipStream_t stream) {
  (void)in_sizes; (void)n_in; (void)out_size; (void)ws_size;
  const float* x  = (const float*)d_in[0];
  const float* Wq = (const float*)d_in[1];
  const float* Wk = (const float*)d_in[2];
  const float* Wv = (const float*)d_in[3];
  const float* bq = (const float*)d_in[4];
  const float* bk = (const float*)d_in[5];
  const float* bv = (const float*)d_in[6];
  float* out = (float*)d_out;

  const size_t PER = (size_t)NB * NH * SEQ * DHEAD; // 6,291,456 elems (bf16)
  __bf16* Qb = (__bf16*)d_ws;
  __bf16* Kb = Qb + PER;
  __bf16* Vt = Qb + 2 * PER; // total ws use: 3*PER*2 = ~37.7 MB

  const int total_waves = NB * NH * (SEQ / 16); // 6144
  const int blocks = total_waves / 8;           // 768 blocks of 8 waves

  qkv_proj_kernel<<<blocks, 256, 0, stream>>>(x, Wq, Wk, Wv, bq, bk, bv,
                                              Qb, Kb, Vt);
  attn_kernel<<<blocks, 256, 0, stream>>>(Qb, Kb, Vt, out);
}
